// SNN_Perceptron_Equalizer_88648124989533
// MI455X (gfx1250) — matrix operand order repaired and checked by reading, not compiled, not run on hardware
//
#include <hip/hip_runtime.h>

// ---- model dims ----
constexpr int TSTEPS = 32;
constexpr int BATCH  = 4096;
constexpr int FH     = 41;    // taps
constexpr int MF     = 164;   // M*F feedforward K
constexpr int MFP    = 192;   // K padded to multiple of 32
constexpr int HID    = 256;
constexpr int NOUT   = 16;
constexpr int XE_STRIDE = 200;  // shorts; 400B rows, 16B aligned, bank-friendly
constexpr int Z_STRIDE  = 264;  // shorts; 528B rows
constexpr int WR_STRIDE = 264;  // LDS W_rec row stride (shorts)
constexpr int WI_STRIDE = 200;  // LDS W_in  row stride (shorts)

typedef __attribute__((ext_vector_type(16))) __bf16         v16bf;
typedef __attribute__((ext_vector_type(8)))  float          v8f;
typedef __attribute__((ext_vector_type(8)))  unsigned short ush8;

union BFrag { v16bf v; ush8 h[2]; };

__device__ __forceinline__ unsigned short f2bf(float f) {
  unsigned int u = __float_as_uint(f);
  u += 0x7FFFu + ((u >> 16) & 1u);            // round-to-nearest-even
  return (unsigned short)(u >> 16);
}

// A-matrix 16x32 bf16 fragment from row-major LDS tile (this code's contiguous K order, not the ISA table order):
// lane row m=lane&15; lanes<16 hold K [k0..k0+15], lanes>=16 [k0+16..k0+31].
__device__ __forceinline__ v16bf load_a_lds(const unsigned short* base, int stride,
                                            int k0, int lane) {
  int m    = lane & 15;
  int koff = (lane & 16) ? 16 : 0;
  const unsigned short* p = base + m * stride + k0 + koff;
  BFrag t;
  t.h[0] = *(const ush8*)(p);
  t.h[1] = *(const ush8*)(p + 8);
  return t.v;
}

// B-matrix 32x16 bf16 fragment from row-major (N x K) LDS weights:
// lane col n=n0+(lane&15); lanes<16 hold K [k0..k0+15], lanes>=16 [k0+16..k0+31].
__device__ __forceinline__ v16bf load_b_lds(const unsigned short* base, int stride,
                                            int n0, int k0, int lane) {
  int n  = n0 + (lane & 15);
  int kb = (lane & 16) ? 16 : 0;
  const unsigned short* p = base + n * stride + k0 + kb;
  BFrag t;
  t.h[0] = *(const ush8*)(p);
  t.h[1] = *(const ush8*)(p + 8);
  return t.v;
}

__device__ __forceinline__ v16bf load_b_glb(const unsigned short* __restrict__ W,
                                            int kstride, int n0, int k0, int lane) {
  int n  = n0 + (lane & 15);
  int kb = (lane & 16) ? 16 : 0;
  const unsigned short* p = W + n * kstride + k0 + kb;
  BFrag t;
  t.h[0] = *(const ush8*)(p);
  t.h[1] = *(const ush8*)(p + 8);
  return t.v;
}

// ---- phase 0: bf16 weight staging + counter reset ----
__global__ void snn_prep(const float* __restrict__ W_in, const float* __restrict__ W_rec,
                         const float* __restrict__ W_out,
                         unsigned short* __restrict__ Winbf,
                         unsigned short* __restrict__ Wrecbf,
                         unsigned short* __restrict__ Woutbf,
                         unsigned int* __restrict__ g_cnt) {
  int i = blockIdx.x * blockDim.x + threadIdx.x;
  if (i == 0) *g_cnt = 0u;
  if (i < HID * MFP) {
    int h = i / MFP, k = i - h * MFP;
    Winbf[i] = (k < MF) ? f2bf(W_in[h * MF + k]) : (unsigned short)0;
  } else if (i < HID * MFP + HID * HID) {
    int j = i - HID * MFP;
    Wrecbf[j] = f2bf(W_rec[j]);
  } else if (i < HID * MFP + HID * HID + NOUT * HID) {
    int j = i - HID * MFP - HID * HID;
    Woutbf[j] = f2bf(W_out[j]);
  }
}

// ---- main persistent LIF kernel ----
// 256 threads = 8 waves = 2 row-groups x 4 column-slot waves.
// Row-group g owns 16 batch rows; wave slot s owns hidden n-tiles [4s..4s+3].
__global__ __launch_bounds__(256) void snn_main(
    const int* __restrict__ x, const float* __restrict__ emb,
    const float* __restrict__ b_out,
    const unsigned short* __restrict__ Winbf,
    const unsigned short* __restrict__ Wrecbf,
    const unsigned short* __restrict__ Woutbf,
    unsigned int* __restrict__ g_cnt,
    float* __restrict__ out) {
  // weights staged once per block via async-to-LDS DMA (padded rows)
  __shared__ __attribute__((aligned(16))) unsigned short s_Wrec[HID * WR_STRIDE]; // 132 KB
  __shared__ __attribute__((aligned(16))) unsigned short s_Win[HID * WI_STRIDE];  // 100 KB
  __shared__ __attribute__((aligned(16))) unsigned short s_emb[16 * 128];         //   4 KB
  __shared__ int s_x[2][16][FH];                                                  // 5.2 KB
  __shared__ __attribute__((aligned(16))) unsigned short s_xe[2][16 * XE_STRIDE]; // 12.5 KB
  __shared__ __attribute__((aligned(16))) unsigned short s_z[2][16 * Z_STRIDE];   // 16.5 KB
  __shared__ unsigned int s_cnt;

  const int tid    = threadIdx.x;
  const int w      = tid >> 5;
  const int lane   = tid & 31;
  const int g      = w >> 2;            // row-group 0/1
  const int s      = w & 3;             // column slot 0..3
  const int NB     = s * 4;             // first n-tile of this wave
  const int blockb = blockIdx.x * 32;
  const int mbase  = 8 * (lane >> 4);   // C-layout: row = mbase + r
  const int ncol   = lane & 15;         // C-layout: col = n*16 + ncol

  // --- async DMA: bf16 weights (global) -> padded LDS images ---
  {
    unsigned lb = (unsigned)(size_t)&s_Wrec[0];
    unsigned long long gb = (unsigned long long)(size_t)Wrecbf;
    for (int c = tid; c < (HID * HID * 2) / 16; c += 256) {   // 8192 x 16B chunks
      int h = c >> 5, cc = c & 31;
      unsigned           laddr = lb + (unsigned)(h * (WR_STRIDE * 2) + cc * 16);
      unsigned long long ga    = gb + (unsigned long long)(h * (HID * 2) + cc * 16);
      asm volatile("global_load_async_to_lds_b128 %0, %1, off"
                   :: "v"(laddr), "v"(ga) : "memory");
    }
    lb = (unsigned)(size_t)&s_Win[0];
    gb = (unsigned long long)(size_t)Winbf;
    for (int c = tid; c < (HID * MFP * 2) / 16; c += 256) {   // 6144 x 16B chunks
      int h = c / 24, cc = c - h * 24;
      unsigned           laddr = lb + (unsigned)(h * (WI_STRIDE * 2) + cc * 16);
      unsigned long long ga    = gb + (unsigned long long)(h * (MFP * 2) + cc * 16);
      asm volatile("global_load_async_to_lds_b128 %0, %1, off"
                   :: "v"(laddr), "v"(ga) : "memory");
    }
  }

  // --- regular init: embedding (bf16), symbols, zeroed z / xe pad ---
  for (int i = tid; i < 16 * 128; i += 256) s_emb[i] = f2bf(emb[i]);
  for (int i = tid; i < 32 * FH; i += 256) {
    int bl = i / FH, f = i - bl * FH;
    s_x[bl >> 4][bl & 15][f] = x[(blockb + bl) * FH + f];
  }
  for (int i = tid; i < 2 * 16 * XE_STRIDE; i += 256) ((unsigned short*)s_xe)[i] = 0;
  for (int i = tid; i < 2 * 16 * Z_STRIDE; i += 256)  ((unsigned short*)s_z)[i]  = 0;
  if (tid == 0) s_cnt = 0u;

  asm volatile("s_wait_asynccnt 0x0" ::: "memory");
  __syncthreads();

  v8f acc[4], vmem[4], zs[4];   // i-current, membrane, spike-sum (C layout, 4 n-tiles)
  #pragma unroll
  for (int q = 0; q < 4; ++q)
    #pragma unroll
    for (int r = 0; r < 8; ++r) { acc[q][r] = 0.f; vmem[q][r] = 0.f; zs[q][r] = 0.f; }

  unsigned int cnt = 0;
  unsigned short* zp  = &s_z[g][0];
  unsigned short* xep = &s_xe[g][0];

  for (int t = 0; t < TSTEPS; ++t) {
    // gather xe(t): this wave fills rows 4s..4s+3 of the group's tile
    for (int m = NB; m < NB + 4; ++m) {
      for (int j = lane; j < MF; j += 32) {
        int flat = t * MF + j;
        int f = flat >> 7, c = flat & 127;
        int sym = s_x[g][m][f];
        xep[m * XE_STRIDE + j] = s_emb[sym * 128 + c];
      }
    }

    // LIF on old i (acc) / v; pack z_new bits; decay i
    unsigned int zb = 0u;
    #pragma unroll
    for (int q = 0; q < 4; ++q) {
      #pragma unroll
      for (int r = 0; r < 8; ++r) {
        float vd = 0.9f * vmem[q][r] + 0.1f * acc[q][r];
        int z = vd > 1.0f;
        vmem[q][r] = z ? 0.0f : vd;
        zs[q][r]  += (float)z;
        cnt       += (unsigned)z;
        zb        |= ((unsigned)z) << ((q << 3) + r);
        acc[q][r] *= 0.8f;
      }
    }

    // recurrent GEMM: i += z_old @ W_rec^T (z_old of whole group, from LDS)
    for (int k0 = 0; k0 < HID; k0 += 32) {
      v16bf a = load_a_lds(zp, Z_STRIDE, k0, lane);
      #pragma unroll
      for (int q = 0; q < 4; ++q) {
        v16bf b = load_b_lds(s_Wrec, WR_STRIDE, (NB + q) * 16, k0, lane);
        acc[q] = __builtin_amdgcn_wmma_f32_16x16x32_bf16(
            false, a, false, b, (short)0, acc[q], false, false);
      }
    }
    __syncthreads();   // all z_old reads done; all xe(t) writes done

    // publish z_new (exact bf16 0/1) into this wave's columns
    #pragma unroll
    for (int q = 0; q < 4; ++q) {
      #pragma unroll
      for (int r = 0; r < 8; ++r) {
        unsigned z = (zb >> ((q << 3) + r)) & 1u;
        zp[(mbase + r) * Z_STRIDE + (NB + q) * 16 + ncol] =
            z ? (unsigned short)0x3F80u : (unsigned short)0u;
      }
    }
    __syncthreads();   // z_new + xe(t) visible to all waves of the group

    // feed-forward GEMM: i += xe_t @ W_in^T
    for (int k0 = 0; k0 < MFP; k0 += 32) {
      v16bf a = load_a_lds(xep, XE_STRIDE, k0, lane);
      #pragma unroll
      for (int q = 0; q < 4; ++q) {
        v16bf b = load_b_lds(s_Win, WI_STRIDE, (NB + q) * 16, k0, lane);
        acc[q] = __builtin_amdgcn_wmma_f32_16x16x32_bf16(
            false, a, false, b, (short)0, acc[q], false, false);
      }
    }
    __syncthreads();   // ff reads of xe(t) done before next gather overwrites
  }

  // decode: out = zsum @ W_out^T + T*b_out   (zsum <= 32, exact in bf16)
  #pragma unroll
  for (int q = 0; q < 4; ++q)
    #pragma unroll
    for (int r = 0; r < 8; ++r)
      zp[(mbase + r) * Z_STRIDE + (NB + q) * 16 + ncol] = f2bf(zs[q][r]);
  __syncthreads();

  if (s == 0) {  // one wave per row-group runs the 256->16 decode GEMM
    v8f od;
    #pragma unroll
    for (int r = 0; r < 8; ++r) od[r] = 0.f;
    for (int k0 = 0; k0 < HID; k0 += 32) {
      v16bf a = load_a_lds(zp, Z_STRIDE, k0, lane);
      v16bf b = load_b_glb(Woutbf, HID, 0, k0, lane);
      od = __builtin_amdgcn_wmma_f32_16x16x32_bf16(
          false, a, false, b, (short)0, od, false, false);
    }
    float bo = (float)TSTEPS * b_out[ncol];
    const int b0 = blockb + g * 16;
    #pragma unroll
    for (int r = 0; r < 8; ++r)
      out[(b0 + mbase + r) * NOUT + ncol] = od[r] + bo;
  }

  // deterministic integer spike counting
  atomicAdd(&s_cnt, cnt);
  __syncthreads();
  if (tid == 0) atomicAdd(g_cnt, s_cnt);
}

// ---- finalize spikerate ----
__global__ void snn_fin(const unsigned int* __restrict__ g_cnt, float* __restrict__ out) {
  if (threadIdx.x == 0 && blockIdx.x == 0)
    out[BATCH * NOUT] = (float)(*g_cnt) * (1.0f / 33554432.0f);  // /(T*B*H)=2^-25
}

extern "C" void kernel_launch(void* const* d_in, const int* in_sizes, int n_in,
                              void* d_out, int out_size, void* d_ws, size_t ws_size,
                              hipStream_t stream) {
  const int*   x     = (const int*)d_in[0];
  const float* emb   = (const float*)d_in[1];
  const float* W_in  = (const float*)d_in[2];
  const float* W_rec = (const float*)d_in[3];
  const float* W_out = (const float*)d_in[4];
  const float* b_out = (const float*)d_in[5];
  float* out = (float*)d_out;

  char* ws = (char*)d_ws;
  unsigned short* Winbf  = (unsigned short*)(ws);                       //  96 KB
  unsigned short* Wrecbf = (unsigned short*)(ws + 98304);               // 128 KB
  unsigned short* Woutbf = (unsigned short*)(ws + 98304 + 131072);      //   8 KB
  unsigned int*   g_cnt  = (unsigned int*)(ws + 98304 + 131072 + 8192); //   4 B

  // stage bf16 weights + reset counter: 49152+65536+4096 = 118784 elems
  snn_prep<<<464, 256, 0, stream>>>(W_in, W_rec, W_out, Winbf, Wrecbf, Woutbf, g_cnt);
  // 128 blocks x 8 waves; 2 row-groups of 16 batch rows, 4 column-slot waves each
  snn_main<<<BATCH / 32, 256, 0, stream>>>(x, emb, b_out, Winbf, Wrecbf, Woutbf, g_cnt, out);
  snn_fin<<<1, 32, 0, stream>>>(g_cnt, out);
}
